// RETAIN_35287451303944
// MI455X (gfx1250) — matrix operand order repaired and checked by reading, not compiled, not run on hardware
//
#include <hip/hip_runtime.h>

// ---------------------------------------------------------------------------
// RETAIN forward for MI455X (gfx1250): all GEMMs via v_wmma_f32_16x16x32_bf16;
// the two 17-GFLOP GEMMs stage tiles into LDS with global_load_async_to_lds_b128
// (ASYNCcnt double-buffering).
// ---------------------------------------------------------------------------

#define B_  32
#define T_  64
#define V_  8192
#define H_  512
#define BT_ (B_ * T_)
#define H3_ (3 * H_)

typedef __attribute__((ext_vector_type(16))) __bf16 bf16x16;
typedef __attribute__((ext_vector_type(8)))  __bf16 bf16x8;
typedef __attribute__((ext_vector_type(8)))  float  f32x8;

__device__ __forceinline__ __bf16 f2bf(float f) {
    union { float f; unsigned u; } v; v.f = f;
    unsigned r = v.u + 0x7FFFu + ((v.u >> 16) & 1u);   // round-to-nearest-even
    union { unsigned short s; __bf16 b; } o;
    o.s = (unsigned short)(r >> 16);
    return o.b;
}

__device__ __forceinline__ float sigmoidf_(float x) {
    return 1.0f / (1.0f + expf(-x));
}

// ---- WMMA fragment loaders -------------------------------------------------
// A (16x32 bf16, row-major, lda elems): lane<16 holds row m0+l, K {0..7,16..23};
// lane>=16 same row set, K {8..15,24..31}.
__device__ __forceinline__ bf16x16 load_a_frag(const __bf16* A, int lda,
                                               int m0, int k0, int lane) {
    int lrow = lane & 15;
    int sel  = lane >> 4;
    const __bf16* p = A + (size_t)(m0 + lrow) * lda + k0 + sel * 8;
    bf16x16 f;
    ((bf16x8*)&f)[0] = *(const bf16x8*)(p);
    ((bf16x8*)&f)[1] = *(const bf16x8*)(p + 16);
    return f;
}

// B (32x16 bf16) from Bt stored [N,K] row-major: lane<16 -> col n0+l, K {0..7,16..23};
// lane>=16 -> K {8..15,24..31}.
__device__ __forceinline__ bf16x16 load_b_frag(const __bf16* Bt, int ldb,
                                               int n0, int k0, int lane) {
    int ncol = lane & 15;
    int sel  = lane >> 4;
    const __bf16* p = Bt + (size_t)(n0 + ncol) * ldb + k0 + sel * 8;
    bf16x16 f;
    ((bf16x8*)&f)[0] = *(const bf16x8*)(p);
    ((bf16x8*)&f)[1] = *(const bf16x8*)(p + 16);
    return f;
}

__device__ __forceinline__ f32x8 wmma_bf16(bf16x16 a, bf16x16 b, f32x8 c) {
    return __builtin_amdgcn_wmma_f32_16x16x32_bf16(false, a, false, b,
                                                   (short)0, c, false, false);
}

// ---- CDNA5 async copy: global -> LDS, tracked by ASYNCcnt ------------------
// lds_off is the wave-relative LDS byte address (low 32 bits of a flat shared
// pointer per the gfx1250 aperture mapping).
__device__ __forceinline__ void async_ld_b128(unsigned lds_off, const void* gaddr) {
    asm volatile("global_load_async_to_lds_b128 %0, %1, off"
                 :: "v"(lds_off), "v"((unsigned long long)(uintptr_t)gaddr)
                 : "memory");
}
__device__ __forceinline__ void wait_async0() {
    asm volatile("s_wait_asynccnt 0" ::: "memory");
}
__device__ __forceinline__ unsigned lds_off_of(const void* p) {
    return (unsigned)(uintptr_t)p;
}

// ---- f32 -> bf16 conversion ------------------------------------------------
__global__ void cvt_bf16_kernel(const float* __restrict__ in,
                                __bf16* __restrict__ out, size_t n) {
    size_t i = (size_t)blockIdx.x * blockDim.x + threadIdx.x;
    size_t stride = (size_t)gridDim.x * blockDim.x;
    for (; i < n; i += stride) out[i] = f2bf(in[i]);
}

// ---- Generic direct-load WMMA GEMM: C[M,N] = epi(A[M,K] @ Bt[N,K]^T) -------
// 256 threads = 8 waves; block tile 64(M) x 128(N); wave tile 32x32 (2x2 WMMA).
enum { EPI_F32_BF16 = 0, EPI_BIAS = 1, EPI_BIAS_SIG = 2, EPI_VALS = 3 };

template <int MODE>
__device__ __forceinline__ void gemm_epilogue(f32x8 acc[2][2], int m0, int n0,
                                              int N, int lane,
                                              float* Cf, __bf16* Cbf,
                                              const float* bias, const float* ve,
                                              const int* lengths, __bf16* valsT) {
    int lcol  = lane & 15;
    int rbase = (lane >> 4) * 8;
    for (int ii = 0; ii < 2; ++ii)
        for (int jj = 0; jj < 2; ++jj)
            for (int r = 0; r < 8; ++r) {
                int row = m0 + ii * 16 + rbase + r;
                int col = n0 + jj * 16 + lcol;
                float v = acc[ii][jj][r];
                size_t idx = (size_t)row * N + col;
                if (MODE == EPI_F32_BF16) {
                    Cf[idx]  = v;
                    Cbf[idx] = f2bf(v);
                } else if (MODE == EPI_BIAS) {
                    Cf[idx] = v + bias[col];
                } else if (MODE == EPI_BIAS_SIG) {
                    Cf[idx] = sigmoidf_(v + bias[col]);
                } else { // EPI_VALS
                    float bt = tanhf(v + bias[col]);
                    int b = row >> 6, t = row & 63;
                    float msk = (t < lengths[b]) ? 1.0f : 0.0f;
                    float out = bt * ve[idx] * msk;
                    valsT[((size_t)b * N + col) * T_ + t] = f2bf(out);
                }
            }
}

template <int MODE>
__launch_bounds__(256)
__global__ void gemm_bf16_kernel(const __bf16* __restrict__ A,
                                 const __bf16* __restrict__ Bt,
                                 int M, int N, int K,
                                 float* __restrict__ Cf,
                                 __bf16* __restrict__ Cbf,
                                 const float* __restrict__ bias,
                                 const float* __restrict__ ve,
                                 const int* __restrict__ lengths,
                                 __bf16* __restrict__ valsT) {
    int lane = threadIdx.x & 31;
    int wave = threadIdx.x >> 5;       // 0..7
    int wy = wave >> 2;                // 0..1
    int wx = wave & 3;                 // 0..3
    int m0 = blockIdx.y * 64 + wy * 32;
    int n0 = blockIdx.x * 128 + wx * 32;

    f32x8 acc[2][2] = {};
    for (int k0 = 0; k0 < K; k0 += 32) {
        if (k0 + 32 < K) {
            __builtin_prefetch(Bt + (size_t)(n0 + (lane & 15)) * K + k0 + 32, 0, 1);
        }
        bf16x16 a0 = load_a_frag(A, K, m0,      k0, lane);
        bf16x16 a1 = load_a_frag(A, K, m0 + 16, k0, lane);
        bf16x16 b0 = load_b_frag(Bt, K, n0,      k0, lane);
        bf16x16 b1 = load_b_frag(Bt, K, n0 + 16, k0, lane);
        acc[0][0] = wmma_bf16(a0, b0, acc[0][0]);
        acc[0][1] = wmma_bf16(a0, b1, acc[0][1]);
        acc[1][0] = wmma_bf16(a1, b0, acc[1][0]);
        acc[1][1] = wmma_bf16(a1, b1, acc[1][1]);
    }
    gemm_epilogue<MODE>(acc, m0, n0, N, lane, Cf, Cbf, bias, ve, lengths, valsT);
}

// ---- LDS-staged WMMA GEMM (big GEMMs): async global->LDS double buffering --
// Per K-step: A tile 64x32 (4KB) + B tile 128x32 (8KB) pulled with
// global_load_async_to_lds_b128 into the ping buffer while computing the pong.
template <int MODE>
__launch_bounds__(256)
__global__ void gemm_bf16_async_kernel(const __bf16* __restrict__ A,
                                       const __bf16* __restrict__ Bt,
                                       int M, int N, int K,
                                       float* __restrict__ Cf,
                                       __bf16* __restrict__ Cbf,
                                       const float* __restrict__ bias) {
    __shared__ __align__(16) __bf16 As[2][64 * 32];    // 8 KB
    __shared__ __align__(16) __bf16 Bs[2][128 * 32];   // 16 KB

    int tid  = threadIdx.x;
    int lane = tid & 31;
    int wave = tid >> 5;
    int wy = wave >> 2, wx = wave & 3;
    int mblk = blockIdx.y * 64;
    int nblk = blockIdx.x * 128;
    int m0l = wy * 32;                 // local row within block tile
    int n0l = wx * 32;                 // local col within block tile

    // staging assignments: 1 b128 of A + 2 b128 of B per thread per step
    int rowA = tid >> 2, segA = tid & 3;                       // 64 x 4
    int idxB0 = tid * 2;
    int rowB0 = idxB0 >> 2, segB0 = idxB0 & 3;                 // 128 x 4
    int rowB1 = (idxB0 + 1) >> 2, segB1 = (idxB0 + 1) & 3;

    auto issue = [&](int k0, int buf) {
        async_ld_b128(lds_off_of(&As[buf][rowA * 32 + segA * 8]),
                      A + (size_t)(mblk + rowA) * K + k0 + segA * 8);
        async_ld_b128(lds_off_of(&Bs[buf][rowB0 * 32 + segB0 * 8]),
                      Bt + (size_t)(nblk + rowB0) * K + k0 + segB0 * 8);
        async_ld_b128(lds_off_of(&Bs[buf][rowB1 * 32 + segB1 * 8]),
                      Bt + (size_t)(nblk + rowB1) * K + k0 + segB1 * 8);
    };

    issue(0, 0);
    wait_async0();
    __syncthreads();

    f32x8 acc[2][2] = {};
    int cur = 0;
    for (int k0 = 0; k0 < K; k0 += 32, cur ^= 1) {
        if (k0 + 32 < K) issue(k0 + 32, cur ^ 1);

        bf16x16 a0 = load_a_frag(&As[cur][0], 32, m0l,      0, lane);
        bf16x16 a1 = load_a_frag(&As[cur][0], 32, m0l + 16, 0, lane);
        bf16x16 b0 = load_b_frag(&Bs[cur][0], 32, n0l,      0, lane);
        bf16x16 b1 = load_b_frag(&Bs[cur][0], 32, n0l + 16, 0, lane);
        acc[0][0] = wmma_bf16(a0, b0, acc[0][0]);
        acc[0][1] = wmma_bf16(a0, b1, acc[0][1]);
        acc[1][0] = wmma_bf16(a1, b0, acc[1][0]);
        acc[1][1] = wmma_bf16(a1, b1, acc[1][1]);

        wait_async0();        // my transfers into buf cur^1 are done
        __syncthreads();      // everyone's transfers done / reads of cur done
    }
    gemm_epilogue<MODE>(acc, mblk + m0l, nblk + n0l, N, lane,
                        Cf, Cbf, bias, nullptr, nullptr, nullptr);
}

// ---- Dual-GRU kernel: blockIdx.x==0 -> GRU a, ==1 -> GRU b -----------------
// 1024 threads = 32 waves. LDS: h f32 [32][512], h bf16 [32][512], gh f32 [32][1536].
__launch_bounds__(1024)
__global__ void gru_kernel(const float* __restrict__ xproj_a,
                           const float* __restrict__ xproj_b,
                           const __bf16* __restrict__ Whh_a,
                           const __bf16* __restrict__ Whh_b,
                           const float* __restrict__ bhh_a,
                           const float* __restrict__ bhh_b,
                           const int* __restrict__ lengths,
                           float* __restrict__ outA_f32,
                           __bf16* __restrict__ outA_bf,
                           float* __restrict__ outB_f32,
                           __bf16* __restrict__ outB_bf) {
    extern __shared__ char smem[];
    float*  h    = (float*)smem;                    // 32*512 f32
    __bf16* h_bf = (__bf16*)(h + B_ * H_);          // 32*512 bf16
    float*  gh   = (float*)(h_bf + B_ * H_);        // 32*1536 f32

    const bool isA = (blockIdx.x == 0);
    const float*  xp  = isA ? xproj_a : xproj_b;
    const __bf16* Whh = isA ? Whh_a   : Whh_b;
    const float*  bhh = isA ? bhh_a   : bhh_b;
    float*  outF = isA ? outA_f32 : outB_f32;
    __bf16* outH = isA ? outA_bf  : outB_bf;

    int tid  = threadIdx.x;
    int lane = tid & 31;
    int wave = tid >> 5;          // 0..31

    for (int e = tid; e < B_ * H_; e += 1024) { h[e] = 0.0f; h_bf[e] = f2bf(0.0f); }
    __syncthreads();

    for (int t = 0; t < T_; ++t) {
        // gh[32,1536] = h[32,512] @ Whh[1536,512]^T  -- 192 WMMA tiles / 32 waves
        for (int i = 0; i < 6; ++i) {
            int tile = wave + 32 * i;        // 0..191
            int tm = tile / 96;              // 0..1
            int tn = tile % 96;              // 0..95
            f32x8 acc = {};
            for (int ks = 0; ks < 16; ++ks) {
                bf16x16 af = load_a_frag(h_bf, H_, tm * 16, ks * 32, lane);
                bf16x16 bf = load_b_frag(Whh, H_, tn * 16, ks * 32, lane);
                acc = wmma_bf16(af, bf, acc);
            }
            int lcol  = lane & 15;
            int rbase = (lane >> 4) * 8;
            for (int r = 0; r < 8; ++r)
                gh[(tm * 16 + rbase + r) * H3_ + tn * 16 + lcol] = acc[r];
        }
        __syncthreads();

        // element-wise gate math; each thread owns 16 h elements
        for (int e = tid; e < B_ * H_; e += 1024) {
            int bb = e >> 9;          // /512
            int j  = e & (H_ - 1);
            size_t xrow = ((size_t)(bb * T_ + t)) * H3_;
            float xr = xp[xrow + j];
            float xz = xp[xrow + H_ + j];
            float xn = xp[xrow + 2 * H_ + j];
            float hr = gh[bb * H3_ + j]           + bhh[j];
            float hz = gh[bb * H3_ + H_ + j]      + bhh[H_ + j];
            float hn = gh[bb * H3_ + 2 * H_ + j]  + bhh[2 * H_ + j];
            float rg = sigmoidf_(xr + hr);
            float zg = sigmoidf_(xz + hz);
            float ng = tanhf(xn + rg * hn);
            float hnew = (1.0f - zg) * ng + zg * h[e];
            h[e]    = hnew;
            h_bf[e] = f2bf(hnew);
            float msk = (t < lengths[bb]) ? 1.0f : 0.0f;
            size_t orow = ((size_t)(bb * T_ + t)) * H_ + j;
            outF[orow] = hnew * msk;
            outH[orow] = f2bf(hnew * msk);
        }
        __syncthreads();
    }
}

// ---- logits -> masked scores: one wave per (b,t) row -----------------------
__launch_bounds__(256)
__global__ void logits_kernel(const float* __restrict__ out_a,
                              const float* __restrict__ w_a,
                              const float* __restrict__ b_a,
                              const int* __restrict__ lengths,
                              float* __restrict__ scores) {
    int lane = threadIdx.x & 31;
    int row  = blockIdx.x * 8 + (threadIdx.x >> 5);   // 0..2047
    if (row >= BT_) return;
    float s = 0.0f;
    for (int k = lane; k < H_; k += 32) s += out_a[(size_t)row * H_ + k] * w_a[k];
    for (int m = 16; m > 0; m >>= 1) s += __shfl_xor(s, m, 32);
    if (lane == 0) {
        int b = row >> 6, t = row & 63;
        float msk = (t < lengths[b]) ? 0.0f : -99999.0f;
        scores[row] = s + b_a[0] + msk;
    }
}

// ---- prefix softmax -> alpha (bf16); rows o >= true_len-1 zeroed -----------
__launch_bounds__(256)
__global__ void alpha_kernel(const float* __restrict__ scores,
                             const int* __restrict__ lengths,
                             __bf16* __restrict__ alpha_bf) {
    int lane = threadIdx.x & 31;
    int wid  = blockIdx.x * 8 + (threadIdx.x >> 5);   // 0..2047
    if (wid >= B_ * T_) return;
    int b = wid >> 6, o = wid & 63;

    int tl = lengths[lane];                    // B==32==wave lanes
    for (int m = 16; m > 0; m >>= 1) tl = max(tl, __shfl_xor(tl, m, 32));

    int t0 = lane, t1 = lane + 32;
    float x0 = (t0 <= o) ? scores[b * T_ + t0] : -1e30f;
    float x1 = (t1 <= o) ? scores[b * T_ + t1] : -1e30f;
    float mx = fmaxf(x0, x1);
    for (int m = 16; m > 0; m >>= 1) mx = fmaxf(mx, __shfl_xor(mx, m, 32));
    float e0 = expf(x0 - mx), e1 = expf(x1 - mx);
    float sm = e0 + e1;
    for (int m = 16; m > 0; m >>= 1) sm += __shfl_xor(sm, m, 32);
    float inv = 1.0f / sm;
    float keep = (o < tl - 1) ? 1.0f : 0.0f;   // progressive_c row mask
    size_t base = ((size_t)(b * T_ + o)) * T_;
    alpha_bf[base + t0] = f2bf(e0 * inv * keep);
    alpha_bf[base + t1] = f2bf(e1 * inv * keep);
}

// ---- batched context: c[b] = alpha[b][64,64] @ valsT[b][512,64]^T ----------
__launch_bounds__(256)
__global__ void context_kernel(const __bf16* __restrict__ alpha_bf,
                               const __bf16* __restrict__ valsT,
                               __bf16* __restrict__ pc_bf) {
    int b    = blockIdx.x;
    int lane = threadIdx.x & 31;
    int wave = threadIdx.x >> 5;
    const __bf16* A  = alpha_bf + (size_t)b * T_ * T_;
    const __bf16* Bt = valsT    + (size_t)b * H_ * T_;
    for (int i = 0; i < 16; ++i) {
        int tile = wave + 8 * i;        // 0..127
        int mt = tile >> 5;             // 0..3
        int nt = tile & 31;             // 0..31
        f32x8 acc = {};
        acc = wmma_bf16(load_a_frag(A, T_, mt * 16, 0,  lane),
                        load_b_frag(Bt, T_, nt * 16, 0,  lane), acc);
        acc = wmma_bf16(load_a_frag(A, T_, mt * 16, 32, lane),
                        load_b_frag(Bt, T_, nt * 16, 32, lane), acc);
        int lcol  = lane & 15;
        int rbase = (lane >> 4) * 8;
        for (int r = 0; r < 8; ++r) {
            int o   = mt * 16 + rbase + r;
            int col = nt * 16 + lcol;
            pc_bf[((size_t)(b * T_ + o)) * H_ + col] = f2bf(acc[r]);
        }
    }
}

// ---------------------------------------------------------------------------
extern "C" void kernel_launch(void* const* d_in, const int* in_sizes, int n_in,
                              void* d_out, int out_size, void* d_ws, size_t ws_size,
                              hipStream_t stream) {
    const float* input_visits = (const float*)d_in[0];
    const int*   lengths      = (const int*)  d_in[1];
    const float* emb_W        = (const float*)d_in[2];
    const float* W_ih_a       = (const float*)d_in[3];
    const float* W_hh_a       = (const float*)d_in[4];
    const float* b_ih_a       = (const float*)d_in[5];
    const float* b_hh_a       = (const float*)d_in[6];
    const float* W_ih_b       = (const float*)d_in[7];
    const float* W_hh_b       = (const float*)d_in[8];
    const float* b_ih_b       = (const float*)d_in[9];
    const float* b_hh_b       = (const float*)d_in[10];
    const float* w_a          = (const float*)d_in[11];
    const float* b_a          = (const float*)d_in[12];
    const float* W_b          = (const float*)d_in[13];
    const float* b_b          = (const float*)d_in[14];
    const float* W_fc         = (const float*)d_in[15];
    const float* b_fc         = (const float*)d_in[16];
    float* out = (float*)d_out;

    // workspace carve-up (all sizes are multiples of 256B)
    char* p = (char*)d_ws;
    auto take = [&](size_t bytes) { char* q = p; p += bytes; return q; };
    __bf16* embW_bf  = (__bf16*)take((size_t)H_ * V_ * 2);
    __bf16* Wih_a_bf = (__bf16*)take((size_t)H3_ * H_ * 2);
    __bf16* Whh_a_bf = (__bf16*)take((size_t)H3_ * H_ * 2);
    __bf16* Wih_b_bf = (__bf16*)take((size_t)H3_ * H_ * 2);
    __bf16* Whh_b_bf = (__bf16*)take((size_t)H3_ * H_ * 2);
    __bf16* Wb_bf    = (__bf16*)take((size_t)H_ * H_ * 2);
    __bf16* Wfc_bf   = (__bf16*)take((size_t)V_ * H_ * 2);
    float*  ve_f32   = (float*) take((size_t)BT_ * H_ * 4);
    __bf16* ve_bf    = (__bf16*)take((size_t)BT_ * H_ * 2);
    float*  xproj_a  = (float*) take((size_t)BT_ * H3_ * 4);
    float*  xproj_b  = (float*) take((size_t)BT_ * H3_ * 4);
    float*  outA_f32 = (float*) take((size_t)BT_ * H_ * 4);
    __bf16* outA_bf  = (__bf16*)take((size_t)BT_ * H_ * 2);
    float*  outB_f32 = (float*) take((size_t)BT_ * H_ * 4);
    __bf16* outB_bf  = (__bf16*)take((size_t)BT_ * H_ * 2);
    float*  scores   = (float*) take((size_t)BT_ * 4);
    __bf16* alpha_bf = (__bf16*)take((size_t)B_ * T_ * T_ * 2);
    __bf16* valsT_bf = (__bf16*)take((size_t)B_ * H_ * T_ * 2);
    __bf16* pc_bf    = (__bf16*)take((size_t)BT_ * H_ * 2);
    (void)ws_size; (void)in_sizes; (void)n_in;

    // input_visits in bf16 lives in the front half of d_out (fully overwritten
    // by the final GEMM afterwards -> deterministic).
    __bf16* ivis_bf = (__bf16*)d_out;

    // --- stage 0: f32 -> bf16 conversions
    auto cvt = [&](const float* src, __bf16* dst, size_t n) {
        int blocks = (int)min((n + 255) / 256, (size_t)2048);
        cvt_bf16_kernel<<<blocks, 256, 0, stream>>>(src, dst, n);
    };
    cvt(input_visits, ivis_bf,  (size_t)BT_ * V_);
    cvt(emb_W,        embW_bf,  (size_t)H_ * V_);
    cvt(W_ih_a,       Wih_a_bf, (size_t)H3_ * H_);
    cvt(W_hh_a,       Whh_a_bf, (size_t)H3_ * H_);
    cvt(W_ih_b,       Wih_b_bf, (size_t)H3_ * H_);
    cvt(W_hh_b,       Whh_b_bf, (size_t)H3_ * H_);
    cvt(W_b,          Wb_bf,    (size_t)H_ * H_);
    cvt(W_fc,         Wfc_bf,   (size_t)V_ * H_);

    // --- stage 1: visit_emb = ivis @ emb_W^T   [2048,512], K=8192 (async-staged)
    gemm_bf16_async_kernel<EPI_F32_BF16><<<dim3(H_ / 128, BT_ / 64), 256, 0, stream>>>(
        ivis_bf, embW_bf, BT_, H_, V_, ve_f32, ve_bf, nullptr);

    // --- stage 2: x projections  [2048,1536], K=512
    gemm_bf16_kernel<EPI_BIAS><<<dim3(H3_ / 128, BT_ / 64), 256, 0, stream>>>(
        ve_bf, Wih_a_bf, BT_, H3_, H_, xproj_a, nullptr, b_ih_a, nullptr, nullptr, nullptr);
    gemm_bf16_kernel<EPI_BIAS><<<dim3(H3_ / 128, BT_ / 64), 256, 0, stream>>>(
        ve_bf, Wih_b_bf, BT_, H3_, H_, xproj_b, nullptr, b_ih_b, nullptr, nullptr, nullptr);

    // --- stage 3: both GRUs (one workgroup each, h + gates resident in LDS)
    size_t gru_lds = (size_t)B_ * H_ * 4 + (size_t)B_ * H_ * 2 + (size_t)B_ * H3_ * 4;
    gru_kernel<<<2, 1024, gru_lds, stream>>>(xproj_a, xproj_b, Whh_a_bf, Whh_b_bf,
                                             b_hh_a, b_hh_b, lengths,
                                             outA_f32, outA_bf, outB_f32, outB_bf);

    // --- stage 4: attention logits + prefix softmax
    logits_kernel<<<BT_ / 8, 256, 0, stream>>>(outA_f32, w_a, b_a, lengths, scores);
    alpha_kernel<<<BT_ / 8, 256, 0, stream>>>(scores, lengths, alpha_bf);

    // --- stage 5: vals^T = (tanh(out_b@W_b^T+b_b) * ve * valid)^T  [per-b 512x64]
    gemm_bf16_kernel<EPI_VALS><<<dim3(H_ / 128, BT_ / 64), 256, 0, stream>>>(
        outB_bf, Wb_bf, BT_, H_, H_, nullptr, nullptr, b_b, ve_f32, lengths, valsT_bf);

    // --- stage 6: progressive_c = alpha @ vals  (batched 64x64x512)
    context_kernel<<<B_, 256, 0, stream>>>(alpha_bf, valsT_bf, pc_bf);

    // --- stage 7: out = sigmoid(pc @ W_fc^T + b_fc)  [2048,8192], K=512 (async-staged)
    gemm_bf16_async_kernel<EPI_BIAS_SIG><<<dim3(V_ / 128, BT_ / 64), 256, 0, stream>>>(
        pc_bf, Wfc_bf, BT_, V_, H_, out, nullptr, b_fc);
}
